// GCN_pia_44306882625585
// MI455X (gfx1250) — compile-verified, matrix operand order checked by reading, not run on hardware
//
#include <hip/hip_runtime.h>
#include <hip/hip_bf16.h>
#include <math.h>

// ---- problem constants (from reference) ----
#define NNODES     50000
#define NFEAT      512
#define NHID       256
#define NCLS       40
#define NCLS_PAD   64      // pad 40 -> 64 = one 4-tile N group
#define WAVE       32

typedef __attribute__((ext_vector_type(16))) __bf16 v16bf;
typedef __attribute__((ext_vector_type(8)))  __bf16 v8bf;
typedef __attribute__((ext_vector_type(8)))  float  v8f;

// ---------------------------------------------------------------------------
// f32 -> bf16 elementwise convert (grid-stride)
// ---------------------------------------------------------------------------
__global__ void k_cvt_bf16(const float* __restrict__ src, __bf16* __restrict__ dst,
                           size_t n) {
    size_t i = (size_t)blockIdx.x * blockDim.x + threadIdx.x;
    size_t stride = (size_t)gridDim.x * blockDim.x;
    for (; i < n; i += stride) dst[i] = (__bf16)src[i];
}

// relu(h) -> bf16
__global__ void k_relu_bf16(const float* __restrict__ src, __bf16* __restrict__ dst,
                            size_t n) {
    size_t i = (size_t)blockIdx.x * blockDim.x + threadIdx.x;
    size_t stride = (size_t)gridDim.x * blockDim.x;
    for (; i < n; i += stride) dst[i] = (__bf16)fmaxf(src[i], 0.0f);
}

// ---------------------------------------------------------------------------
// W [K x Nc] f32 row-major -> Wt [NcPad x K] bf16 row-major (zero pad rows)
// ---------------------------------------------------------------------------
__global__ void k_transpose_bf16(const float* __restrict__ W, __bf16* __restrict__ Wt,
                                 int K, int Nc, int NcPad) {
    int idx = blockIdx.x * blockDim.x + threadIdx.x;
    int total = NcPad * K;
    if (idx >= total) return;
    int c = idx / K;
    int k = idx - c * K;
    Wt[(size_t)c * K + k] = (c < Nc) ? (__bf16)W[(size_t)k * Nc + c] : (__bf16)0.0f;
}

// h[n][f] = b[f]  (broadcast bias as the segment-sum initial value)
__global__ void k_init_bias(float* __restrict__ h, const float* __restrict__ b,
                            size_t total, int F) {
    size_t i = (size_t)blockIdx.x * blockDim.x + threadIdx.x;
    size_t stride = (size_t)gridDim.x * blockDim.x;
    for (; i < total; i += stride) h[i] = b[i % F];
}

// ---------------------------------------------------------------------------
// WMMA bf16 GEMM, 16x64 tile per wave (4 N-tiles share one A fragment):
//   C[M x ldc] = A[M x K] (bf16 row-major) @ Bt^T,  Bt is [Ncols x K] bf16.
// Per K-step: 1 A-fragment load feeds 4 back-to-back v_wmma_f32_16x16x32_bf16.
// Fragment layout per ISA 7.12.2 (16-bit A 16x32): lane<16 -> K {0..7,16..23},
// lane>=16 -> K {8..15,24..31}; B loaded "column-major" = same pattern on Bt.
// ---------------------------------------------------------------------------
__global__ void __launch_bounds__(256)
k_gemm_bf16_wmma(const __bf16* __restrict__ A, const __bf16* __restrict__ Bt,
                 float* __restrict__ C, int Ngroups, int Twaves, int K, int ldc) {
    int g = blockIdx.x * 8 + (threadIdx.x >> 5);   // global wave id
    if (g >= Twaves) return;                       // uniform per wave
    int lane   = threadIdx.x & (WAVE - 1);
    int grp    = g % Ngroups;                      // 64-wide N group
    int tile_m = g / Ngroups;
    int ln     = lane & 15;
    int koff   = (lane >> 4) * 8;                  // half-wave K offset

    const __bf16* arow = A  + (size_t)(tile_m * 16 + ln) * K + koff;
    const __bf16* brow = Bt + (size_t)(grp * 64 + ln) * K + koff;
    const size_t  bstr = (size_t)16 * K;           // next 16 B-columns

    union U { v16bf v; v8bf h[2]; };
    v8f acc0 = {}, acc1 = {}, acc2 = {}, acc3 = {};
    for (int k0 = 0; k0 < K; k0 += 32) {
        __builtin_prefetch(arow + k0 + 128, 0, 1);   // global_prefetch_b8
        U ua, ub0, ub1, ub2, ub3;
        ua.h[0]  = *(const v8bf*)(arow + k0);
        ua.h[1]  = *(const v8bf*)(arow + k0 + 16);
        ub0.h[0] = *(const v8bf*)(brow + k0);
        ub0.h[1] = *(const v8bf*)(brow + k0 + 16);
        ub1.h[0] = *(const v8bf*)(brow + bstr + k0);
        ub1.h[1] = *(const v8bf*)(brow + bstr + k0 + 16);
        ub2.h[0] = *(const v8bf*)(brow + 2 * bstr + k0);
        ub2.h[1] = *(const v8bf*)(brow + 2 * bstr + k0 + 16);
        ub3.h[0] = *(const v8bf*)(brow + 3 * bstr + k0);
        ub3.h[1] = *(const v8bf*)(brow + 3 * bstr + k0 + 16);
        acc0 = __builtin_amdgcn_wmma_f32_16x16x32_bf16(
                   false, ua.v, false, ub0.v, (short)0, acc0, false, false);
        acc1 = __builtin_amdgcn_wmma_f32_16x16x32_bf16(
                   false, ua.v, false, ub1.v, (short)0, acc1, false, false);
        acc2 = __builtin_amdgcn_wmma_f32_16x16x32_bf16(
                   false, ua.v, false, ub2.v, (short)0, acc2, false, false);
        acc3 = __builtin_amdgcn_wmma_f32_16x16x32_bf16(
                   false, ua.v, false, ub3.v, (short)0, acc3, false, false);
    }

    // C/D layout: VGPR v -> M = v + 8*(lane>=16), N = lane%16
    int ncol  = grp * 64 + ln;
    int mbase = tile_m * 16 + (lane >> 4) * 8;
#pragma unroll
    for (int v = 0; v < 8; ++v) {
        float* crow = C + (size_t)(mbase + v) * ldc + ncol;
        crow[0]  = acc0[v];
        crow[16] = acc1[v];
        crow[32] = acc2[v];
        crow[48] = acc3[v];
    }
}

// ---------------------------------------------------------------------------
// SpMM: for edge e, feature f:  h[dst[e]][f] += support[src[e]][f] * w[e]
// one block per edge, one thread per feature; f32 hardware atomics (L2-resident)
// ---------------------------------------------------------------------------
__global__ void k_spmm_edges(const float* __restrict__ sup,
                             const int* __restrict__ src, const int* __restrict__ dst,
                             const float* __restrict__ w, float* __restrict__ h,
                             int F, int ldsup) {
    int e = blockIdx.x;
    int f = threadIdx.x;
    if (f >= F) return;
    int s = src[e], d = dst[e];
    float v = sup[(size_t)s * ldsup + f] * w[e];
    unsafeAtomicAdd(&h[(size_t)d * F + f], v);     // global_atomic_add_f32
}

// ---------------------------------------------------------------------------
// log-softmax over 40 classes, one thread per node
// ---------------------------------------------------------------------------
__global__ void k_logsoftmax(const float* __restrict__ h2, float* __restrict__ logp,
                             int n_nodes) {
    int n = blockIdx.x * blockDim.x + threadIdx.x;
    if (n >= n_nodes) return;
    const float* r = h2 + (size_t)n * NCLS;
    float m = -INFINITY;
#pragma unroll 8
    for (int f = 0; f < NCLS; ++f) m = fmaxf(m, r[f]);
    float s = 0.0f;
#pragma unroll 8
    for (int f = 0; f < NCLS; ++f) s += expf(r[f] - m);
    float lse = m + logf(s);
    float* o = logp + (size_t)n * NCLS;
#pragma unroll 8
    for (int f = 0; f < NCLS; ++f) o[f] = r[f] - lse;
}

// ---------------------------------------------------------------------------
extern "C" void kernel_launch(void* const* d_in, const int* in_sizes, int n_in,
                              void* d_out, int out_size, void* d_ws, size_t ws_size,
                              hipStream_t stream) {
    const float* x        = (const float*)d_in[0];
    const int*   edge_src = (const int*)  d_in[1];
    const int*   edge_dst = (const int*)  d_in[2];
    const float* edge_w   = (const float*)d_in[3];
    const float* W1       = (const float*)d_in[4];
    const float* b1       = (const float*)d_in[5];
    const float* W2       = (const float*)d_in[6];
    const float* b2       = (const float*)d_in[7];

    const int N = in_sizes[0] / NFEAT;   // 50000
    const int E = in_sizes[1];           // 800000

    // ---- output partition: (logp, h1, h2) concatenated flat ----
    float* out  = (float*)d_out;
    float* logp = out;
    float* h1   = out + (size_t)N * NCLS;
    float* h2   = out + (size_t)N * NCLS + (size_t)N * NHID;

    // ---- workspace layout (bytes) ----
    char* ws = (char*)d_ws;
    size_t off = 0;
    __bf16* xb   = (__bf16*)(ws + off); off += (size_t)N * NFEAT * sizeof(__bf16);       // 51.2 MB
    __bf16* W1t  = (__bf16*)(ws + off); off += (size_t)NHID * NFEAT * sizeof(__bf16);    // 256 KB
    float*  sup1 = (float*) (ws + off); off += (size_t)N * NHID * sizeof(float);         // 51.2 MB
    __bf16* a1b  = (__bf16*)(ws + off); off += (size_t)N * NHID * sizeof(__bf16);        // 25.6 MB
    __bf16* W2t  = (__bf16*)(ws + off); off += (size_t)NCLS_PAD * NHID * sizeof(__bf16); // 32 KB
    float*  sup2 = (float*) (ws + off); off += (size_t)N * NCLS_PAD * sizeof(float);     // 12.8 MB
    (void)ws_size;

    // 1) x -> bf16
    {
        size_t n = (size_t)N * NFEAT;
        k_cvt_bf16<<<(int)((n + 255) / 256), 256, 0, stream>>>(x, xb, n);
    }
    // 2) W1 [512x256] -> W1t [256x512] bf16 ; W2 [256x40] -> W2t [64x256] bf16
    {
        int t1 = NHID * NFEAT;
        k_transpose_bf16<<<(t1 + 255) / 256, 256, 0, stream>>>(W1, W1t, NFEAT, NHID, NHID);
        int t2 = NCLS_PAD * NHID;
        k_transpose_bf16<<<(t2 + 255) / 256, 256, 0, stream>>>(W2, W2t, NHID, NCLS, NCLS_PAD);
    }
    // 3) h1 = b1, h2 = b2 (segment-sum initial values)
    {
        size_t t1 = (size_t)N * NHID;
        k_init_bias<<<(int)((t1 + 255) / 256), 256, 0, stream>>>(h1, b1, t1, NHID);
        size_t t2 = (size_t)N * NCLS;
        k_init_bias<<<(int)((t2 + 255) / 256), 256, 0, stream>>>(h2, b2, t2, NCLS);
    }
    // 4) GEMM1: sup1[N x 256] = xb[N x 512] @ W1   (16x64 per wave)
    {
        int Mtiles  = N / 16;              // 3125 (exact)
        int Ngroups = NHID / 64;           // 4
        int T = Mtiles * Ngroups;          // 12500 waves
        k_gemm_bf16_wmma<<<(T + 7) / 8, 256, 0, stream>>>(xb, W1t, sup1, Ngroups, T, NFEAT, NHID);
    }
    // 5) SpMM1: h1 += scatter-add of sup1[src]*w
    k_spmm_edges<<<E, 256, 0, stream>>>(sup1, edge_src, edge_dst, edge_w, h1, NHID, NHID);
    // 6) a1 = relu(h1) -> bf16
    {
        size_t n = (size_t)N * NHID;
        k_relu_bf16<<<(int)((n + 255) / 256), 256, 0, stream>>>(h1, a1b, n);
    }
    // 7) GEMM2: sup2[N x 64] = a1b[N x 256] @ W2 (padded)
    {
        int Mtiles  = N / 16;              // 3125
        int Ngroups = NCLS_PAD / 64;       // 1
        int T = Mtiles * Ngroups;          // 3125 waves
        k_gemm_bf16_wmma<<<(T + 7) / 8, 256, 0, stream>>>(a1b, W2t, sup2, Ngroups, T, NHID, NCLS_PAD);
    }
    // 8) SpMM2: h2 += scatter-add of sup2[src]*w (first 40 cols only)
    k_spmm_edges<<<E, 64, 0, stream>>>(sup2, edge_src, edge_dst, edge_w, h2, NCLS, NCLS_PAD);
    // 9) logp = log_softmax(h2)
    k_logsoftmax<<<(N + 255) / 256, 256, 0, stream>>>(h2, logp, N);
}
